// MultiHeadCrossAttention_76450417869263
// MI455X (gfx1250) — compile-verified, matrix operand order checked by reading, not run on hardware
//
#include <hip/hip_runtime.h>
#include <hip/hip_bf16.h>

// ---------------------------------------------------------------------------
// ProbSparse cross-attention for MI455X (gfx1250, wave32, WMMA).
// All heavy math runs through v_wmma_f32_16x16x32_bf16, 2x4-register-blocked:
// each wave owns a 32x64 C block -> 8 WMMAs per 6 fragment loads
// (21.3 FLOP/byte of L2 traffic; 2x2 gave 16, naive single-tile gave 8).
// ---------------------------------------------------------------------------

typedef __bf16 bf16_t;
typedef __attribute__((ext_vector_type(8)))  bf16_t v8bf;
typedef __attribute__((ext_vector_type(16))) bf16_t v16bf;
typedef __attribute__((ext_vector_type(8)))  float  v8f;

#define B_      4
#define L_      2048
#define D_      512
#define H_      8
#define DH_     64
#define NTOP    40
#define NTOPP   64          // padded to 4 x 16 so every GEMM dim is % 32
#define SCALE_  0.125f      // 1/sqrt(64)

__device__ __forceinline__ unsigned short f2bf(float f) {
  union { float f; unsigned u; } c; c.f = f;
  unsigned u = c.u;
  unsigned r = (u + 0x7FFFu + ((u >> 16) & 1u)) >> 16;   // RNE
  return (unsigned short)r;
}
__device__ __forceinline__ float bf2f(unsigned short h) {
  union { unsigned u; float f; } c; c.u = ((unsigned)h) << 16; return c.f;
}

// -------------------------------- convert ---------------------------------
__global__ void k_f32_to_bf16(const float* __restrict__ in,
                              unsigned short* __restrict__ out, int n) {
  int i = blockIdx.x * blockDim.x + threadIdx.x;
  if (i < n) out[i] = f2bf(in[i]);
}

// ---------------- batched WMMA GEMM: C = alpha*(A·Bᵀ) + bias ---------------
// A: [M,K] bf16 row-major, B: [N,K] bf16 row-major, C: [M,N] f32.
// 2x4 register blocking: one 32x64 C block per wave, 8 waves per 256-thr block.
// Fragment layout per CDNA5 ISA §7.12.2:
//   A 16x32: lane<16 holds row (lane&15), K = {0..7,16..23}; lane>=16 holds
//   K = {8..15,24..31}. B 32x16 mirrors it with lane = output column.
//   C/D f32: lane = column (lane&15), rows (lane>>4)*8 + v, v=0..7.
__device__ __forceinline__ v16bf load_frag(const unsigned short* p) {
  v8bf lo = *(const v8bf*)(p);
  v8bf hi = *(const v8bf*)(p + 16);
  return __builtin_shufflevector(lo, hi, 0,1,2,3,4,5,6,7,8,9,10,11,12,13,14,15);
}

#define WMMA_BF16(ACC, AV, BV) \
  ACC = __builtin_amdgcn_wmma_f32_16x16x32_bf16(false, AV, false, BV, (short)0, ACC, false, false)

__global__ void k_wmma_gemm_bt(const unsigned short* __restrict__ A,
                               const unsigned short* __restrict__ Bm,
                               const float* __restrict__ bias,
                               float* __restrict__ C,
                               int M, int N, int K,
                               long sA, long sB, long sC, float alpha) {
  const int batch = blockIdx.z;
  A  += (long)batch * sA;
  Bm += (long)batch * sB;
  C  += (long)batch * sC;

  const int lane = threadIdx.x & 31;
  const int wave = threadIdx.x >> 5;
  const int tn4 = N >> 6, tm2 = M >> 5;            // 32x64 macro-tiles
  const int tile = blockIdx.x * 8 + wave;
  if (tile >= tm2 * tn4) return;                   // wave-uniform, EXEC full
  const int m0 = (tile / tn4) << 5;
  const int n0 = (tile % tn4) << 6;

  const int r  = lane & 15;
  const int kh = (lane >> 4) << 3;                 // 0 or 8

  const unsigned short* ap0 = A  + (long)(m0 + r)      * K + kh;
  const unsigned short* ap1 = A  + (long)(m0 + 16 + r) * K + kh;
  const unsigned short* bp0 = Bm + (long)(n0 + r)      * K + kh;
  const unsigned short* bp1 = Bm + (long)(n0 + 16 + r) * K + kh;
  const unsigned short* bp2 = Bm + (long)(n0 + 32 + r) * K + kh;
  const unsigned short* bp3 = Bm + (long)(n0 + 48 + r) * K + kh;

  v8f acc00 = {}, acc01 = {}, acc02 = {}, acc03 = {};
  v8f acc10 = {}, acc11 = {}, acc12 = {}, acc13 = {};
  for (int k0 = 0; k0 < K; k0 += 32) {
    v16bf a0 = load_frag(ap0);
    v16bf a1 = load_frag(ap1);
    v16bf b0 = load_frag(bp0);
    v16bf b1 = load_frag(bp1);
    v16bf b2 = load_frag(bp2);
    v16bf b3 = load_frag(bp3);
    __builtin_prefetch(ap0 + 32, 0, 3);            // global_prefetch_b8
    __builtin_prefetch(bp0 + 32, 0, 3);
    WMMA_BF16(acc00, a0, b0);
    WMMA_BF16(acc10, a1, b0);
    WMMA_BF16(acc01, a0, b1);
    WMMA_BF16(acc11, a1, b1);
    WMMA_BF16(acc02, a0, b2);
    WMMA_BF16(acc12, a1, b2);
    WMMA_BF16(acc03, a0, b3);
    WMMA_BF16(acc13, a1, b3);
    ap0 += 32; ap1 += 32;
    bp0 += 32; bp1 += 32; bp2 += 32; bp3 += 32;
  }

  const int mbA = m0 + kh, mbB = m0 + 16 + kh;
  const int n_0 = n0 + r, n_1 = n0 + 16 + r, n_2 = n0 + 32 + r, n_3 = n0 + 48 + r;
  const float b0s = bias ? bias[n_0] : 0.0f;
  const float b1s = bias ? bias[n_1] : 0.0f;
  const float b2s = bias ? bias[n_2] : 0.0f;
  const float b3s = bias ? bias[n_3] : 0.0f;
#pragma unroll
  for (int v = 0; v < 8; v++) {
    float* rowA = C + (long)(mbA + v) * N;
    float* rowB = C + (long)(mbB + v) * N;
    rowA[n_0] = alpha * acc00[v] + b0s;
    rowA[n_1] = alpha * acc01[v] + b1s;
    rowA[n_2] = alpha * acc02[v] + b2s;
    rowA[n_3] = alpha * acc03[v] + b3s;
    rowB[n_0] = alpha * acc10[v] + b0s;
    rowB[n_1] = alpha * acc11[v] + b1s;
    rowB[n_2] = alpha * acc12[v] + b2s;
    rowB[n_3] = alpha * acc13[v] + b3s;
  }
}

// ---------------------- head-repack helpers (bandwidth-trivial) ------------
__global__ void k_repack_qk(const float* __restrict__ Xf,
                            unsigned short* __restrict__ Xb, int total) {
  int i = blockIdx.x * blockDim.x + threadIdx.x;
  if (i >= total) return;
  int n = i % D_; long bl = i / D_;
  int l = (int)(bl % L_); int b = (int)(bl / L_);
  int h = n >> 6, dh = n & 63;
  Xb[(((long)(b * H_ + h) * L_ + l) << 6) + dh] = f2bf(Xf[i]);
}

__global__ void k_repack_vt(const float* __restrict__ Vf,
                            unsigned short* __restrict__ Vtb, int total) {
  int i = blockIdx.x * blockDim.x + threadIdx.x;
  if (i >= total) return;
  int n = i % D_; long bl = i / D_;
  int l = (int)(bl % L_); int b = (int)(bl / L_);
  int h = n >> 6, dh = n & 63;
  Vtb[((long)(b * H_ + h) * DH_ + dh) * L_ + l] = f2bf(Vf[i]);  // V^T per head
}

// ---------------------------- mean(V) over L -------------------------------
__global__ void k_vmean(const float* __restrict__ Vf, float* __restrict__ Vm) {
  int bh = blockIdx.x, d = threadIdx.x;            // blockDim = 64
  int b = bh / H_, h = bh % H_;
  const float* p = Vf + (long)b * L_ * D_ + h * DH_ + d;
  float s = 0.f;
  for (int l = 0; l < L_; l++) s += p[(long)l * D_];
  Vm[bh * DH_ + d] = s * (1.0f / L_);
}

// ---------------- sparsity measure M = max(QKs) - sum(QKs)/L_enc -----------
__global__ void k_msample(const unsigned short* __restrict__ Qb,
                          const unsigned short* __restrict__ Kb,
                          const int* __restrict__ idxs, int sample_k,
                          float* __restrict__ Msc) {
  int i = blockIdx.x * blockDim.x + threadIdx.x;    // over B*H*L
  if (i >= B_ * H_ * L_) return;
  int l = i % L_, bh = i / L_;
  const unsigned short* q = Qb + ((long)bh * L_ + l) * DH_;
  float qf[DH_];
  for (int d = 0; d < DH_; d++) qf[d] = bf2f(q[d]);
  float mx = -3.4e38f, sm = 0.f;
  for (int s = 0; s < sample_k; s++) {
    int ks = idxs[l * sample_k + s];
    const unsigned short* kr = Kb + ((long)bh * L_ + ks) * DH_;
    float acc = 0.f;
    for (int d = 0; d < DH_; d++) acc += qf[d] * bf2f(kr[d]);
    mx = fmaxf(mx, acc); sm += acc;
  }
  Msc[i] = mx - sm * (1.0f / L_);
}

// ------------------------ iterative top-k (40 of 2048) ---------------------
__global__ void k_topk(const float* __restrict__ Msc, int* __restrict__ topidx) {
  __shared__ float sM[L_];
  __shared__ float rv[256];
  __shared__ int   ri[256];
  int bh = blockIdx.x, tid = threadIdx.x;
  for (int i = tid; i < L_; i += 256) sM[i] = Msc[(long)bh * L_ + i];
  __syncthreads();
  for (int t = 0; t < NTOP; t++) {
    float bv = -3.4e38f; int bi = 0;
    for (int i = tid; i < L_; i += 256) { float v = sM[i]; if (v > bv) { bv = v; bi = i; } }
    rv[tid] = bv; ri[tid] = bi; __syncthreads();
    for (int s = 128; s > 0; s >>= 1) {
      if (tid < s && rv[tid + s] > rv[tid]) { rv[tid] = rv[tid + s]; ri[tid] = ri[tid + s]; }
      __syncthreads();
    }
    if (tid == 0) { topidx[bh * NTOPP + t] = ri[0]; sM[ri[0]] = -3.4e38f; }
    __syncthreads();
  }
}

__global__ void k_gather_qred(const unsigned short* __restrict__ Qb,
                              const int* __restrict__ topidx,
                              unsigned short* __restrict__ Qred) {
  int bh = blockIdx.x / NTOPP, u = blockIdx.x % NTOPP, d = threadIdx.x; // 64
  unsigned short v = 0;                                   // zero-pad rows 40..63
  if (u < NTOP) {
    int l = topidx[bh * NTOPP + u];
    v = Qb[((long)bh * L_ + l) * DH_ + d];
  }
  Qred[((long)bh * NTOPP + u) * DH_ + d] = v;
}

// ------------------------------- softmax row -------------------------------
__global__ void k_softmax(const float* __restrict__ scores,
                          unsigned short* __restrict__ attn) {
  __shared__ float red[256];
  int row = blockIdx.x, tid = threadIdx.x;
  const float* p = scores + (long)row * L_;
  float mx = -3.4e38f;
  for (int i = tid; i < L_; i += 256) mx = fmaxf(mx, p[i]);
  red[tid] = mx; __syncthreads();
  for (int s = 128; s > 0; s >>= 1) { if (tid < s) red[tid] = fmaxf(red[tid], red[tid + s]); __syncthreads(); }
  mx = red[0]; __syncthreads();
  float sm = 0.f;
  for (int i = tid; i < L_; i += 256) sm += __expf(p[i] - mx);
  red[tid] = sm; __syncthreads();
  for (int s = 128; s > 0; s >>= 1) { if (tid < s) red[tid] += red[tid + s]; __syncthreads(); }
  float inv = 1.0f / red[0];
  unsigned short* o = attn + (long)row * L_;
  for (int i = tid; i < L_; i += 256) o[i] = f2bf(__expf(p[i] - mx) * inv);
}

// --------------------- ctx = broadcast mean(V), then scatter ---------------
__global__ void k_ctx_fill(const float* __restrict__ Vm,
                           unsigned short* __restrict__ ctxb, int total) {
  int i = blockIdx.x * blockDim.x + threadIdx.x;
  if (i >= total) return;
  int n = i % D_; long bl = i / D_; int b = (int)(bl / L_);
  ctxb[i] = f2bf(Vm[(b * H_ + (n >> 6)) * DH_ + (n & 63)]);
}

__global__ void k_scatter(const float* __restrict__ updf,
                          const int* __restrict__ topidx,
                          unsigned short* __restrict__ ctxb) {
  int bh = blockIdx.x / NTOP, u = blockIdx.x % NTOP, d = threadIdx.x; // 64
  int b = bh / H_, h = bh % H_;
  int l = topidx[bh * NTOPP + u];
  ctxb[((long)(b * L_ + l)) * D_ + h * DH_ + d] =
      f2bf(updf[((long)bh * NTOPP + u) * DH_ + d]);
}

// ---------------------------------------------------------------------------
extern "C" void kernel_launch(void* const* d_in, const int* in_sizes, int n_in,
                              void* d_out, int out_size, void* d_ws, size_t ws_size,
                              hipStream_t stream) {
  const float* x    = (const float*)d_in[0];
  const float* ctx  = (const float*)d_in[1];
  const float* Wq   = (const float*)d_in[2];
  const float* bq   = (const float*)d_in[3];
  const float* Wk   = (const float*)d_in[4];
  const float* bk   = (const float*)d_in[5];
  const float* Wv   = (const float*)d_in[6];
  const float* bv   = (const float*)d_in[7];
  const float* Wo   = (const float*)d_in[8];
  const float* bo   = (const float*)d_in[9];
  const int*   smp  = (const int*)d_in[10];
  float*       out  = (float*)d_out;

  const int SX  = B_ * L_ * D_;                 // 4,194,304
  const int SW  = D_ * D_;                      // 262,144
  const int sample_k = in_sizes[10] / L_;       // 40 for this setup

  // -------- workspace arena (bytes; large f32 buffers reused later) --------
  char* w = (char*)d_ws;
  size_t off = 0;
  auto take = [&](size_t bytes) { char* p = w + off; off = (off + bytes + 255) & ~(size_t)255; return p; };
  unsigned short* xb   = (unsigned short*)take((size_t)SX * 2);
  unsigned short* cb   = (unsigned short*)take((size_t)SX * 2);
  unsigned short* wqb  = (unsigned short*)take((size_t)SW * 2);
  unsigned short* wkb  = (unsigned short*)take((size_t)SW * 2);
  unsigned short* wvb  = (unsigned short*)take((size_t)SW * 2);
  unsigned short* wob  = (unsigned short*)take((size_t)SW * 2);
  float*          Qf   = (float*)take((size_t)SX * 4);   // reused as scores
  float*          Kf   = (float*)take((size_t)SX * 4);   // reused as attn
  float*          Vf   = (float*)take((size_t)SX * 4);   // reused as ctx_bf16
  unsigned short* Qb   = (unsigned short*)take((size_t)SX * 2);
  unsigned short* Kb   = (unsigned short*)take((size_t)SX * 2);
  unsigned short* Vtb  = (unsigned short*)take((size_t)SX * 2);
  float*          Vm   = (float*)take((size_t)B_ * H_ * DH_ * 4);
  float*          Msc  = (float*)take((size_t)B_ * H_ * L_ * 4);
  int*            tidx = (int*)take((size_t)B_ * H_ * NTOPP * 4);
  unsigned short* Qred = (unsigned short*)take((size_t)B_ * H_ * NTOPP * DH_ * 2);
  float*          updf = (float*)take((size_t)B_ * H_ * NTOPP * DH_ * 4);
  float*          scores = Qf;                          // 32*64*2048 f32 == 16MB
  unsigned short* attnb  = (unsigned short*)Kf;         // 32*64*2048 bf16 == 8MB
  unsigned short* ctxb   = (unsigned short*)Vf;         // B*L*D bf16 == 8MB
  (void)ws_size; (void)n_in; (void)out_size;

  // 1) f32 -> bf16 conversions
  k_f32_to_bf16<<<(SX + 255) / 256, 256, 0, stream>>>(x,   xb,  SX);
  k_f32_to_bf16<<<(SX + 255) / 256, 256, 0, stream>>>(ctx, cb,  SX);
  k_f32_to_bf16<<<(SW + 255) / 256, 256, 0, stream>>>(Wq,  wqb, SW);
  k_f32_to_bf16<<<(SW + 255) / 256, 256, 0, stream>>>(Wk,  wkb, SW);
  k_f32_to_bf16<<<(SW + 255) / 256, 256, 0, stream>>>(Wv,  wvb, SW);
  k_f32_to_bf16<<<(SW + 255) / 256, 256, 0, stream>>>(Wo,  wob, SW);

  // 2) Q/K/V projections: [8192,512] = [8192,512] x [512,512]^T (WMMA)
  {
    const int M = B_ * L_, N = D_, K = D_;
    dim3 grid(((M / 32) * (N / 64) + 7) / 8, 1, 1);
    k_wmma_gemm_bt<<<grid, 256, 0, stream>>>(xb, wqb, bq, Qf, M, N, K, 0, 0, 0, 1.0f);
    k_wmma_gemm_bt<<<grid, 256, 0, stream>>>(cb, wkb, bk, Kf, M, N, K, 0, 0, 0, 1.0f);
    k_wmma_gemm_bt<<<grid, 256, 0, stream>>>(cb, wvb, bv, Vf, M, N, K, 0, 0, 0, 1.0f);
  }

  // 3) repack to per-head layouts
  k_repack_qk<<<(SX + 255) / 256, 256, 0, stream>>>(Qf, Qb, SX);
  k_repack_qk<<<(SX + 255) / 256, 256, 0, stream>>>(Kf, Kb, SX);
  k_repack_vt<<<(SX + 255) / 256, 256, 0, stream>>>(Vf, Vtb, SX);

  // 4) mean(V) over L per (b,h,d)
  k_vmean<<<B_ * H_, DH_, 0, stream>>>(Vf, Vm);

  // 5) sampled sparsity measure M
  k_msample<<<(B_ * H_ * L_ + 127) / 128, 128, 0, stream>>>(Qb, Kb, smp, sample_k, Msc);

  // 6) top-k + gather Q_red (zero-padded to 64 rows)
  k_topk<<<B_ * H_, 256, 0, stream>>>(Msc, tidx);
  k_gather_qred<<<B_ * H_ * NTOPP, DH_, 0, stream>>>(Qb, tidx, Qred);

  // 7) scores = SCALE * Q_red x K^T : batched WMMA [64,2048] per (b,h)
  {
    dim3 grid(((NTOPP / 32) * (L_ / 64) + 7) / 8, 1, B_ * H_);
    k_wmma_gemm_bt<<<grid, 256, 0, stream>>>(Qred, Kb, nullptr, scores,
        NTOPP, L_, DH_,
        (long)NTOPP * DH_, (long)L_ * DH_, (long)NTOPP * L_, SCALE_);
  }

  // 8) softmax rows -> bf16 attn
  k_softmax<<<B_ * H_ * NTOPP, 256, 0, stream>>>(scores, attnb);

  // 9) upd = attn x V : batched WMMA [64,64] per (b,h), K=2048, B = V^T rows
  {
    dim3 grid(((NTOPP / 32) * (DH_ / 64) + 7) / 8, 1, B_ * H_);
    k_wmma_gemm_bt<<<grid, 256, 0, stream>>>(attnb, Vtb, nullptr, updf,
        NTOPP, DH_, L_,
        (long)NTOPP * L_, (long)DH_ * L_, (long)NTOPP * DH_, 1.0f);
  }

  // 10) ctx = broadcast mean(V); scatter upd rows; output projection
  k_ctx_fill<<<(SX + 255) / 256, 256, 0, stream>>>(Vm, ctxb, SX);
  k_scatter<<<B_ * H_ * NTOP, DH_, 0, stream>>>(updf, tidx, ctxb);
  {
    const int M = B_ * L_, N = D_, K = D_;
    dim3 grid(((M / 32) * (N / 64) + 7) / 8, 1, 1);
    k_wmma_gemm_bt<<<grid, 256, 0, stream>>>(ctxb, wob, bo, out, M, N, K, 0, 0, 0, 1.0f);
  }
}
